// Strong_ConvCIFAR10_KAN_Generator_33921651704114
// MI455X (gfx1250) — compile-verified, matrix operand order checked by reading, not run on hardware
//
#include <hip/hip_runtime.h>
#include <hip/hip_bf16.h>

typedef __attribute__((ext_vector_type(16))) __bf16 v16bf;
typedef __attribute__((ext_vector_type(8)))  float  v8f;

union FragU { unsigned int u[8]; v16bf v; };

// ---------- helpers ----------

__device__ __forceinline__ unsigned short f2bf(float f) {
    union { float f; unsigned int u; } x;
    x.f = f;
    unsigned int u = x.u;
    unsigned int r = u + 0x7FFFu + ((u >> 16) & 1u);   // round-to-nearest-even
    return (unsigned short)(r >> 16);
}

// Load a 16x32 bf16 fragment from an LDS tile (row stride in elements).
// lane&15 = row, lane>>4 selects K half-split: VGPR0..3 = K kb..kb+7,
// VGPR4..7 = K kb+16..kb+23.
__device__ __forceinline__ void load_frag(const unsigned short* base, int row0,
                                          int kofs, int rowstride, FragU& f) {
    int lane = threadIdx.x & 31;
    int r    = lane & 15;
    int kb   = kofs + ((lane >> 4) << 3);
    const unsigned short* p = base + (size_t)(row0 + r) * rowstride + kb;
    const unsigned int* p32 = (const unsigned int*)p;          // 4B aligned
    const unsigned int* q32 = (const unsigned int*)(p + 16);
#pragma unroll
    for (int j = 0; j < 4; ++j) f.u[j] = p32[j];
#pragma unroll
    for (int j = 0; j < 4; ++j) f.u[4 + j] = q32[j];
}

// ---------- kernel 1: linear + relu, write NHWC [B,2,2,512] ----------

__global__ __launch_bounds__(256) void linear_relu_kernel(
    const float* __restrict__ x, const float* __restrict__ w,
    const float* __restrict__ bias, float* __restrict__ h0) {
    int idx = blockIdx.x * blockDim.x + threadIdx.x;   // 64*2048
    if (idx >= 64 * 2048) return;
    int b = idx >> 11;
    int f = idx & 2047;
    float acc = bias[f];
    const float* xr = x + b * 100;
    const float* wr = w + f * 100;
#pragma unroll 4
    for (int k = 0; k < 100; ++k) acc = fmaf(xr[k], wr[k], acc);
    acc = fmaxf(acc, 0.0f);
    // reshape [B,512,2,2]: f = c*4 + y*2 + xq ; store NHWC [B,2,2,512]
    int c = f >> 2, y = (f >> 1) & 1, xq = f & 1;
    h0[((size_t)(b * 2 + y) * 2 + xq) * 512 + c] = acc;
}

// ---------- kernel 2: pack B matrix to bf16 [CoutPad, fin*9] ----------

__global__ __launch_bounds__(256) void pack_b_kernel(
    const float* __restrict__ bw, const float* __restrict__ sw,
    const float* __restrict__ ss, unsigned short* __restrict__ out,
    int Cout, int CoutPad, int fin) {
    int K = fin * 9;
    int total = CoutPad * K;
    int idx = blockIdx.x * blockDim.x + threadIdx.x;
    if (idx >= total) return;
    int o = idx / K;
    int k = idx - o * K;
    int i = k / 9;
    int comp = k - i * 9;
    float v = 0.0f;
    if (o < Cout) {
        if (comp == 0) v = bw[(size_t)o * fin + i];
        else v = sw[((size_t)o * fin + i) * 8 + (comp - 1)] * ss[(size_t)o * fin + i];
    }
    out[idx] = f2bf(v);
}

// ---------- kernel 3: fused expand(B-splines) + bf16 WMMA GEMM ----------
// prev: NHWC [B, Hp, Wp, Cin] (pre-upsample); conv runs at H = 2*Hp, W = 2*Wp.
// Block tile: 32 rows (pixels) x 64 cols (out channels); 8 waves as 2(M)x4(N).
// K-tile = 288 bf16 = 32 input scalars x 9 components = 9 WMMA steps of K=32.

#define ASTRIDE 296   // 288 + pad: 148 dwords -> 16B-aligned rows, conflict-free

__global__ __launch_bounds__(256) void kan_gemm_kernel(
    const float* __restrict__ prev, const unsigned short* __restrict__ Bp,
    float* __restrict__ outAct, float* __restrict__ outImg,
    int Hp, int Wp, int Cin, int H, int W, int Cout, int CoutPad, int finalMode) {

    const int fin    = Cin * 9;
    const int K      = fin * 9;
    const int nTiles = K / 288;

    __shared__ unsigned short sA[32 * ASTRIDE];
    __shared__ unsigned short sB[64 * ASTRIDE];

    const int t    = threadIdx.x;
    const int lane = t & 31;
    const int wave = t >> 5;
    const int wm16 = (wave >> 2) * 16;  // M sub-tile row (0/16)
    const int wn16 = (wave & 3) * 16;   // N sub-tile col (0..48)
    const int n0   = blockIdx.x * 32;
    const int o0   = blockIdx.y * 64;
    const int HW   = H * W;

    // ---- kt-invariant per-thread expansion state (4 scalars/thread/K-tile) ----
    int cA[4], tpA[4], rbA[4], pbA[4], yA[4], xA[4];
#pragma unroll
    for (int u = 0; u < 4; ++u) {
        int idx = u * 256 + t;        // 0..1023
        int r   = idx >> 5;           // pixel row in tile (0..31)
        int ii  = idx & 31;           // scalar within K-tile
        int n   = n0 + r;
        int bI  = n / HW;
        int rem = n - bI * HW;
        int y   = rem / W;
        int xq  = rem - y * W;
        cA[u]  = ii / 9;              // channel (advanced incrementally per kt)
        tpA[u] = ii - cA[u] * 9;      // tap (advanced incrementally per kt)
        rbA[u] = r * ASTRIDE + ii * 9;
        pbA[u] = bI * Hp * Wp * Cin;
        yA[u]  = y;
        xA[u]  = xq;
    }

    // ---- affine B staging: thread t copies 36 consecutive dwords.
    // 144 dwords per 288-elem row slice -> row cc = t>>2, quarter (t&3)*36.
    const int ccB = t >> 2;
    const int e0B = (t & 3) * 36;
    const unsigned int* gRowB =
        (const unsigned int*)Bp + (size_t)(o0 + ccB) * (K >> 1) + e0B;
    unsigned int* lRowB = (unsigned int*)sB + ccB * (ASTRIDE / 2) + e0B;

    v8f acc = {};

    for (int kt = 0; kt < nTiles; ++kt) {
        // ---- stage A: 32 rows x 32 scalars, each expanded to 9 bf16 ----
#pragma unroll
        for (int u = 0; u < 4; ++u) {
            int tap = tpA[u], c = cA[u];
            int dy  = (tap >= 6) ? 1 : ((tap >= 3) ? 0 : -1);
            int dx  = tap - (dy + 1) * 3 - 1;
            int yy  = yA[u] + dy;
            int xx  = xA[u] + dx;
            float v = 0.0f;
            if ((unsigned)yy < (unsigned)H && (unsigned)xx < (unsigned)W)
                v = prev[pbA[u] + ((yy >> 1) * Wp + (xx >> 1)) * Cin + c];

            unsigned short* dst = &sA[rbA[u]];
            // silu(x) = x * sigmoid(x)
            dst[0] = f2bf(v / (1.0f + __expf(-v)));
#pragma unroll
            for (int q = 1; q < 9; ++q) dst[q] = 0;
            // cubic B-spline, uniform knots g[j] = (j-3)*0.4 - 1:
            // p = interval, u = frac; nonzero bases at slots p-3..p
            float tt = (v + 2.2f) * 2.5f;
            float fp = floorf(tt);
            int   p  = (int)fp;
            float uu = tt - fp;
            float om = 1.0f - uu;
            float w0 = om * om * om * (1.0f / 6.0f);
            float w3 = uu * uu * uu * (1.0f / 6.0f);
            float w1 = ((3.0f * uu - 6.0f) * uu * uu + 4.0f) * (1.0f / 6.0f);
            float w2v = (((-3.0f * uu + 3.0f) * uu + 3.0f) * uu + 1.0f) * (1.0f / 6.0f);
            int j0 = p - 3;
            if ((unsigned)j0       < 8u) dst[1 + j0] = f2bf(w0);
            if ((unsigned)(j0 + 1) < 8u) dst[2 + j0] = f2bf(w1);
            if ((unsigned)(j0 + 2) < 8u) dst[3 + j0] = f2bf(w2v);
            if ((unsigned)(j0 + 3) < 8u) dst[4 + j0] = f2bf(w3);

            // advance (i += 32): tap += 32 mod 9 = 5, c += 3 (+carry)
            tap += 5; c += 3;
            if (tap >= 9) { tap -= 9; ++c; }
            tpA[u] = tap; cA[u] = c;
        }

        // ---- stage B: 64 cols x 288 bf16 (contiguous 36-dword runs/thread) ----
        {
            const unsigned int* g = gRowB + kt * 144;
#pragma unroll
            for (int w2 = 0; w2 < 36; ++w2) lRowB[w2] = g[w2];
        }

        __syncthreads();

        // ---- 9 WMMA steps of K=32, fragment double-buffered ----
        {
            FragU fa[2], fb[2];
            load_frag(sA, wm16, 0, ASTRIDE, fa[0]);
            load_frag(sB, wn16, 0, ASTRIDE, fb[0]);
#pragma unroll
            for (int s = 0; s < 9; ++s) {
                int cur = s & 1, nxt = cur ^ 1;
                if (s < 8) {
                    load_frag(sA, wm16, (s + 1) * 32, ASTRIDE, fa[nxt]);
                    load_frag(sB, wn16, (s + 1) * 32, ASTRIDE, fb[nxt]);
                }
                acc = __builtin_amdgcn_wmma_f32_16x16x32_bf16(
                    false, fa[cur].v, false, fb[cur].v, (short)0, acc, false, false);
            }
        }

        __syncthreads();
    }

    // ---- store: C/D layout: VGPR j -> M = j + 8*(lane>=16), N = lane&15 ----
    const int nl   = lane & 15;
    const int half = lane >> 4;
#pragma unroll
    for (int j = 0; j < 8; ++j) {
        int m = j + half * 8;
        int n = n0 + wm16 + m;
        int o = o0 + wn16 + nl;
        float v = acc[j];
        if (!finalMode) {
            outAct[(size_t)n * CoutPad + o] = v;
        } else if (o < Cout) {
            int bI  = n / HW;
            int rem = n - bI * HW;
            int y   = rem / W;
            int xq  = rem - y * W;
            outImg[(((size_t)bI * Cout + o) * H + y) * W + xq] = tanhf(v);
        }
    }
}

// ---------- host launch ----------

extern "C" void kernel_launch(void* const* d_in, const int* in_sizes, int n_in,
                              void* d_out, int out_size, void* d_ws, size_t ws_size,
                              hipStream_t stream) {
    (void)in_sizes; (void)n_in; (void)out_size; (void)ws_size;

    const float* x   = (const float*)d_in[0];
    const float* lw  = (const float*)d_in[1];
    const float* lb  = (const float*)d_in[2];
    const float* bw[4] = {(const float*)d_in[3], (const float*)d_in[6],
                          (const float*)d_in[9], (const float*)d_in[12]};
    const float* sw[4] = {(const float*)d_in[4], (const float*)d_in[7],
                          (const float*)d_in[10], (const float*)d_in[13]};
    const float* ss[4] = {(const float*)d_in[5], (const float*)d_in[8],
                          (const float*)d_in[11], (const float*)d_in[14]};

    const int Cin[4]   = {512, 256, 128, 64};
    const int CoutR[4] = {256, 128, 64, 3};
    const int CoutP[4] = {256, 128, 64, 64};   // padded to N-block (64)
    const int Hh[4]    = {4, 8, 16, 32};       // conv resolution per layer

    char* ws = (char*)d_ws;
    size_t off = 0;
    auto alloc = [&](size_t bytes) -> char* {
        char* p = ws + off;
        off = (off + bytes + 255) & ~(size_t)255;
        return p;
    };

    float* h0   = (float*)alloc((size_t)64 * 2048 * 4);
    float* act1 = (float*)alloc((size_t)1024 * 256 * 4);
    float* act2 = (float*)alloc((size_t)4096 * 128 * 4);
    float* act3 = (float*)alloc((size_t)16384 * 64 * 4);
    unsigned short* Bp[4];
    for (int l = 0; l < 4; ++l) {
        size_t Kl = (size_t)Cin[l] * 81;
        Bp[l] = (unsigned short*)alloc((size_t)CoutP[l] * Kl * 2);
    }

    // pack weights to bf16 [CoutPad, K]
    for (int l = 0; l < 4; ++l) {
        int fin = Cin[l] * 9;
        int total = CoutP[l] * fin * 9;
        int blocks = (total + 255) / 256;
        pack_b_kernel<<<blocks, 256, 0, stream>>>(bw[l], sw[l], ss[l], Bp[l],
                                                  CoutR[l], CoutP[l], fin);
    }

    // linear + relu -> NHWC [64,2,2,512]
    linear_relu_kernel<<<(64 * 2048 + 255) / 256, 256, 0, stream>>>(x, lw, lb, h0);

    const float* prevs[4] = {h0, act1, act2, act3};
    float* acts[4]        = {act1, act2, act3, nullptr};

    for (int l = 0; l < 4; ++l) {
        int H = Hh[l], W = Hh[l];
        int Hp = H / 2, Wp = W / 2;
        int M = 64 * H * W;
        dim3 grid(M / 32, CoutP[l] / 64);
        int finalMode = (l == 3) ? 1 : 0;
        kan_gemm_kernel<<<grid, 256, 0, stream>>>(
            prevs[l], Bp[l], acts[l], (float*)d_out,
            Hp, Wp, Cin[l], H, W, CoutR[l], CoutP[l], finalMode);
    }
}
